// VoxelProposalLayerMultiBS_79783312490964
// MI455X (gfx1250) — compile-verified, matrix operand order checked by reading, not run on hardware
//
#include <hip/hip_runtime.h>
#include <hip/hip_bf16.h>
#include <math.h>

typedef __attribute__((ext_vector_type(16))) _Float16 v16h;
typedef __attribute__((ext_vector_type(8)))  float    v8f;

#define XD 128
#define YD 128
#define ZD 8
#define NTOT (XD*YD*ZD)       // 131072 queries per batch
#define CCH 128
#define NHEADS 8
#define NLEV 3
#define NPTS 4
#define STOT 9507             // 47*153 + 24*77 + 12*39
#define BATCH 2
#define RPAD 19024            // BATCH*STOT padded up to multiple of 16

// ---------------------------------------------------------------- WMMA helpers

__device__ __forceinline__ v8f wmma16(v16h a, v16h b, v8f c) {
  // D = A(16x32 f16) x B(32x16 f16) + C(16x16 f32)
  return __builtin_amdgcn_wmma_f32_16x16x32_f16(false, a, false, b, (short)0, c,
                                                false, false);
}

// B fragment from pre-swizzled fragment-ready layout:
// halves stored at [((nt*4 + kt)*32 + lane)*16 + e]  -> one aligned 32B vector load
__device__ __forceinline__ v16h load_bfrag_sw(const _Float16* __restrict__ Wsw,
                                              int nt, int kt, int lane) {
  return *reinterpret_cast<const v16h*>(Wsw + (((nt * 4 + kt) * 32 + lane) << 4));
}

// ---------------------------------------------------------------- weight repack
// Repack f32 weights into f16 *fragment-ready* swizzled layout so the GEMM loop
// loads each B fragment as a single v16h (2x global_load_b128, no v_mov_b16 packing).
// Element mapping (16-bit B 32x16 tile): lane<16 -> N=lane, K runs {0..7,16..23};
// lane>=16 -> N=lane-16, K runs {8..15,24..31}.

__device__ __forceinline__ void frag_coords(int i, int& nt, int& kt, int& n15,
                                            int& K) {
  int e = i & 15, lane = (i >> 4) & 31;
  kt = (i >> 9) & 3;
  nt = i >> 11;
  n15 = lane & 15;
  K = kt * 32 + ((lane & 16) >> 1) + ((e & 8) << 1) + (e & 7);
}

__global__ void k_convert(const float* __restrict__ Woff, const float* __restrict__ Ww,
                          const float* __restrict__ Wv,   const float* __restrict__ Wo,
                          _Float16* __restrict__ WcatS, _Float16* __restrict__ WvS,
                          _Float16* __restrict__ WoS) {
  int i = blockIdx.x * blockDim.x + threadIdx.x;
  const int CATSZ = 18 * 4 * 32 * 16;   // 36864 halves (128x288)
  const int SQSZ  = 8 * 4 * 32 * 16;    // 16384 halves (128x128)
  if (i < CATSZ) {
    int nt, kt, n15, K;
    frag_coords(i, nt, kt, n15, K);
    int n = nt * 16 + n15;
    float v = (n < 192) ? Woff[K * 192 + n] : Ww[K * 96 + (n - 192)];
    WcatS[i] = (_Float16)v;
    return;
  }
  i -= CATSZ;
  if (i < SQSZ) {
    int nt, kt, n15, K;
    frag_coords(i, nt, kt, n15, K);
    WvS[i] = (_Float16)Wv[K * CCH + nt * 16 + n15];
    return;
  }
  i -= SQSZ;
  if (i < SQSZ) {
    int nt, kt, n15, K;
    frag_coords(i, nt, kt, n15, K);
    WoS[i] = (_Float16)Wo[K * CCH + nt * 16 + n15];
  }
}

// ---------------------------------------------------------------- voxel mask

__global__ void k_mask_zero(unsigned char* __restrict__ mask) {
  int i = blockIdx.x * blockDim.x + threadIdx.x;
  if (i < BATCH * NTOT) mask[i] = 0;
}

__global__ void k_mask_scatter(const int* __restrict__ vol,
                               unsigned char* __restrict__ mask) {
  int i = blockIdx.x * blockDim.x + threadIdx.x;
  if (i >= BATCH * NTOT) return;
  int b  = i / NTOT;
  int vx = vol[i * 3 + 0], vy = vol[i * 3 + 1], vz = vol[i * 3 + 2];
  if (vx >= 0 && vx < XD && vy >= 0 && vy < YD && vz >= 0 && vz < ZD)
    mask[b * NTOT + vx * (YD * ZD) + vy * ZD + vz] = 1;  // races benign: all write 1
}

// ---------------------------------------------------------------- value projection
// value[(b*STOT+s), c] = sum_k feat_l[b][k][j] * Wv[k][c] + bv[c]
// value buffer is padded to RPAD rows: the last tile stores clamped-duplicate
// rows unconditionally (never read), so the epilogue has no predicated stores.

__global__ void __launch_bounds__(128) k_value(
    const float* __restrict__ f0, const float* __restrict__ f1,
    const float* __restrict__ f2, const _Float16* __restrict__ WvS,
    const float* __restrict__ bv, float* __restrict__ value) {
  const int wave = threadIdx.x >> 5;
  const int lane = threadIdx.x & 31;
  const int R = BATCH * STOT;
  const int mtiles = RPAD / 16;         // 1189
  const int mtile = blockIdx.x * 4 + wave;
  if (mtile >= mtiles) return;          // wave-uniform, no barriers in this kernel
  const int row0 = mtile * 16;

  // A fragments: row = (b, s); element k strided by HW in the NCHW feat
  v16h af[4];
  {
    int row = min(row0 + (lane & 15), R - 1);
    int bb = row / STOT;
    int s  = row - bb * STOT;
    const float* f; int hw, j;
    if (s < 7191)      { f = f0; hw = 7191; j = s; }
    else if (s < 9039) { f = f1; hw = 1848; j = s - 7191; }
    else               { f = f2; hw = 468;  j = s - 9039; }
    const float* base = f + (size_t)bb * CCH * hw + j;
    const int kofs = (lane & 16) >> 1;
#pragma unroll
    for (int kt = 0; kt < 4; ++kt)
#pragma unroll
      for (int e = 0; e < 16; ++e) {
        int K = kt * 32 + kofs + ((e & 8) << 1) + (e & 7);
        af[kt][e] = (_Float16)base[(size_t)K * hw];
      }
  }
  const int mb  = (lane & 16) ? 8 : 0;
  const int nlo = lane & 15;
#pragma unroll 2
  for (int nt = 0; nt < 8; ++nt) {
    v8f c = {};
#pragma unroll
    for (int kt = 0; kt < 4; ++kt)
      c = wmma16(af[kt], load_bfrag_sw(WvS, nt, kt, lane), c);
    int ncol = nt * 16 + nlo;
    float bias = bv[ncol];
#pragma unroll
    for (int r = 0; r < 8; ++r)
      value[(size_t)(row0 + mb + r) * CCH + ncol] = c[r] + bias;
  }
}

// ---------------------------------------------------------------- fused attention
// per 16-query tile: proj GEMM -> softmax -> bilinear gather -> out GEMM ->
// residual -> LayerNorm -> masked transposed store. One wave per tile.

__global__ void __launch_bounds__(64) k_fused(
    const float* __restrict__ se, const float* __restrict__ sp,
    const float* __restrict__ ref_pix,
    const float* __restrict__ boff, const float* __restrict__ bwgt,
    const float* __restrict__ bo, const float* __restrict__ ln_g,
    const float* __restrict__ ln_b,
    const float* __restrict__ value, const _Float16* __restrict__ WcatS,
    const _Float16* __restrict__ WoS, const unsigned char* __restrict__ mask,
    float* __restrict__ out) {
  __shared__ float projS[2][16][288];   // 192 offsets + 96 logits per query (later: out rows)
  __shared__ float accS[2][16][128];    // sampled features, channel = h*16+d

  const int wave = threadIdx.x >> 5;
  const int lane = threadIdx.x & 31;
  const int mtile = blockIdx.x * 2 + wave;     // grid covers all 16384 tiles exactly
  const int row0 = mtile * 16;
  const int b  = row0 / NTOT;                  // NTOT % 16 == 0 -> uniform per tile
  const int n0 = row0 - b * NTOT;

  // ---- Phase A: q = scene_embed + scene_pos, packed as f16 A fragments
  v16h qa[4];
  {
    const int M = lane & 15;
    const size_t rowBase = (size_t)(row0 + M) * CCH;
    const int kofs = (lane & 16) >> 1;
#pragma unroll
    for (int kt = 0; kt < 4; ++kt) {
#pragma unroll
      for (int r = 0; r < 2; ++r) {
        const int K = kt * 32 + kofs + r * 16;
        const float4* pe = reinterpret_cast<const float4*>(se + rowBase + K);
        const float4* pp = reinterpret_cast<const float4*>(sp + rowBase + K);
        float4 e0 = pe[0], e1 = pe[1], p0 = pp[0], p1 = pp[1];
        qa[kt][r * 8 + 0] = (_Float16)(e0.x + p0.x);
        qa[kt][r * 8 + 1] = (_Float16)(e0.y + p0.y);
        qa[kt][r * 8 + 2] = (_Float16)(e0.z + p0.z);
        qa[kt][r * 8 + 3] = (_Float16)(e0.w + p0.w);
        qa[kt][r * 8 + 4] = (_Float16)(e1.x + p1.x);
        qa[kt][r * 8 + 5] = (_Float16)(e1.y + p1.y);
        qa[kt][r * 8 + 6] = (_Float16)(e1.z + p1.z);
        qa[kt][r * 8 + 7] = (_Float16)(e1.w + p1.w);
      }
    }
  }

  // ---- Phase B: [16x288] = q @ [Woff | Ww]  (18 N-tiles x 4 K-tiles of WMMA)
  {
    const int mb  = (lane & 16) ? 8 : 0;
    const int nlo = lane & 15;
#pragma unroll 2
    for (int nt = 0; nt < 18; ++nt) {
      v8f c = {};
#pragma unroll
      for (int kt = 0; kt < 4; ++kt)
        c = wmma16(qa[kt], load_bfrag_sw(WcatS, nt, kt, lane), c);
      int ncol = nt * 16 + nlo;
      float bias = (ncol < 192) ? boff[ncol] : bwgt[ncol - 192];
#pragma unroll
      for (int r = 0; r < 8; ++r)
        projS[wave][mb + r][ncol] = c[r] + bias;
    }
  }
  __syncthreads();

  // ---- Phase C: softmax + 3-level / 4-point bilinear gather (4 (q,h) pairs/lane)
  {
    const int lw[NLEV]    = {153, 77, 39};
    const int lh[NLEV]    = {47, 24, 12};
    const int lbase[NLEV] = {0, 7191, 9039};
    for (int it = 0; it < 4; ++it) {
      const int pair = it * 32 + lane;
      const int q = pair >> 3;      // 0..15
      const int h = pair & 7;       // 0..7
      float lg[12];
      float mx = -1e30f;
#pragma unroll
      for (int j = 0; j < 12; ++j) {
        lg[j] = projS[wave][q][192 + h * 12 + j];
        mx = fmaxf(mx, lg[j]);
      }
      float ssum = 0.f;
#pragma unroll
      for (int j = 0; j < 12; ++j) { lg[j] = __expf(lg[j] - mx); ssum += lg[j]; }
      const float inv = 1.f / ssum;

      const int node = n0 + q;
      const float rx = ref_pix[((size_t)b * NTOT + node) * 2 + 0];
      const float ry = ref_pix[((size_t)b * NTOT + node) * 2 + 1];

      float4 a0 = {0, 0, 0, 0}, a1 = {0, 0, 0, 0}, a2 = {0, 0, 0, 0}, a3 = {0, 0, 0, 0};
      for (int l = 0; l < NLEV; ++l) {
        const int wl = lw[l], hl = lh[l];
        const float* vlev = value + ((size_t)b * STOT + lbase[l]) * CCH + h * 16;
        for (int pt = 0; pt < NPTS; ++pt) {
          const float offx = projS[wave][q][((h * NLEV + l) * NPTS + pt) * 2 + 0];
          const float offy = projS[wave][q][((h * NLEV + l) * NPTS + pt) * 2 + 1];
          const float watt = lg[l * NPTS + pt] * inv;
          // loc*W - 0.5 with loc = ref + off/W  ->  rx*W + offx - 0.5
          const float xf = rx * (float)wl + offx - 0.5f;
          const float yf = ry * (float)hl + offy - 0.5f;
          const float x0 = floorf(xf), y0 = floorf(yf);
          const float fx = xf - x0, fy = yf - y0;
          const int x0i = (int)x0, y0i = (int)y0;
          const float cw[4] = {(1.f - fx) * (1.f - fy), fx * (1.f - fy),
                               (1.f - fx) * fy, fx * fy};
#pragma unroll
          for (int cor = 0; cor < 4; ++cor) {
            const int xi = x0i + (cor & 1);
            const int yi = y0i + (cor >> 1);
            const bool valid = (xi >= 0) & (xi < wl) & (yi >= 0) & (yi < hl);
            const int xc = min(max(xi, 0), wl - 1);
            const int yc = min(max(yi, 0), hl - 1);
            const float wgt = valid ? cw[cor] * watt : 0.f;
            const float4* vp =
                reinterpret_cast<const float4*>(vlev + (size_t)(yc * wl + xc) * CCH);
            float4 g0 = vp[0], g1 = vp[1], g2 = vp[2], g3 = vp[3];
            a0.x += g0.x * wgt; a0.y += g0.y * wgt; a0.z += g0.z * wgt; a0.w += g0.w * wgt;
            a1.x += g1.x * wgt; a1.y += g1.y * wgt; a1.z += g1.z * wgt; a1.w += g1.w * wgt;
            a2.x += g2.x * wgt; a2.y += g2.y * wgt; a2.z += g2.z * wgt; a2.w += g2.w * wgt;
            a3.x += g3.x * wgt; a3.y += g3.y * wgt; a3.z += g3.z * wgt; a3.w += g3.w * wgt;
          }
        }
      }
      float4* ap = reinterpret_cast<float4*>(&accS[wave][q][h * 16]);
      ap[0] = a0; ap[1] = a1; ap[2] = a2; ap[3] = a3;
    }
  }
  __syncthreads();

  // ---- Phase D: out[16x128] = acc @ Wo + bo + identity
  {
    v16h af[4];
    const int M = lane & 15;
    const int kofs = (lane & 16) >> 1;
#pragma unroll
    for (int kt = 0; kt < 4; ++kt)
#pragma unroll
      for (int e = 0; e < 16; ++e) {
        int K = kt * 32 + kofs + ((e & 8) << 1) + (e & 7);
        af[kt][e] = (_Float16)accS[wave][M][K];
      }
    const int mb  = (lane & 16) ? 8 : 0;
    const int nlo = lane & 15;
#pragma unroll 2
    for (int nt = 0; nt < 8; ++nt) {
      v8f c = {};
#pragma unroll
      for (int kt = 0; kt < 4; ++kt)
        c = wmma16(af[kt], load_bfrag_sw(WoS, nt, kt, lane), c);
      int ncol = nt * 16 + nlo;
      float bias = bo[ncol];
#pragma unroll
      for (int r = 0; r < 8; ++r) {
        int m = mb + r;
        projS[wave][m][ncol] = c[r] + bias + se[(size_t)(row0 + m) * CCH + ncol];
      }
    }
  }
  __syncthreads();

  // ---- Phase E: LayerNorm + mask select + [B,C,N] transposed store.
  // Full-wave: lanes (L, L+16) co-own row L&15; each handles 64 channels,
  // partial sums exchanged via wave32 shuffle (no EXEC divergence).
  {
    const int m    = lane & 15;
    const int c0   = (lane >> 4) * 64;
    float s0 = 0.f;
#pragma unroll 4
    for (int c2 = 0; c2 < 64; ++c2) s0 += projS[wave][m][c0 + c2];
    const float mu = (s0 + __shfl_xor(s0, 16, 32)) * (1.f / CCH);
    float v0 = 0.f;
#pragma unroll 4
    for (int c2 = 0; c2 < 64; ++c2) {
      float d = projS[wave][m][c0 + c2] - mu;
      v0 += d * d;
    }
    const float var  = (v0 + __shfl_xor(v0, 16, 32)) * (1.f / CCH);
    const float rstd = rsqrtf(var + 1e-5f);
    const int node = n0 + m;
    const bool msk = mask[(size_t)b * NTOT + node] != 0;
    const float* idrow = se + (size_t)(row0 + m) * CCH;
#pragma unroll 4
    for (int c2 = c0; c2 < c0 + 64; ++c2) {
      float v = msk ? ((projS[wave][m][c2] - mu) * rstd * ln_g[c2] + ln_b[c2])
                    : idrow[c2];
      out[((size_t)b * CCH + c2) * NTOT + node] = v;
    }
  }
}

// ---------------------------------------------------------------- launch

extern "C" void kernel_launch(void* const* d_in, const int* in_sizes, int n_in,
                              void* d_out, int out_size, void* d_ws, size_t ws_size,
                              hipStream_t stream) {
  const float* se   = (const float*)d_in[0];
  const float* sp   = (const float*)d_in[1];
  const int*   vol  = (const int*)d_in[2];
  const float* refp = (const float*)d_in[3];
  const float* Wv   = (const float*)d_in[4];
  const float* bv   = (const float*)d_in[5];
  const float* Woff = (const float*)d_in[6];
  const float* boff = (const float*)d_in[7];
  const float* Ww   = (const float*)d_in[8];
  const float* bwgt = (const float*)d_in[9];
  const float* Wo   = (const float*)d_in[10];
  const float* bo   = (const float*)d_in[11];
  const float* lng  = (const float*)d_in[12];
  const float* lnb  = (const float*)d_in[13];
  const float* f0   = (const float*)d_in[14];
  const float* f1   = (const float*)d_in[15];
  const float* f2   = (const float*)d_in[16];
  float* out = (float*)d_out;

  // workspace layout (~10.2 MB total)
  char* ws = (char*)d_ws;
  _Float16* WcatS = (_Float16*)(ws);                     // 128*288*2 = 73728
  _Float16* WvS   = (_Float16*)(ws + 73728);             // 32768
  _Float16* WoS   = (_Float16*)(ws + 73728 + 32768);     // 32768
  float* value    = (float*)(ws + 139264);               // RPAD*128*4 = 9740288
  unsigned char* mask =
      (unsigned char*)(ws + 139264 + (size_t)RPAD * CCH * 4);        // 262144

  const int convN = CCH * 288 + 2 * CCH * CCH;           // 69632
  k_convert<<<(convN + 255) / 256, 256, 0, stream>>>(Woff, Ww, Wv, Wo, WcatS, WvS, WoS);
  k_mask_zero<<<(BATCH * NTOT + 255) / 256, 256, 0, stream>>>(mask);
  k_mask_scatter<<<(BATCH * NTOT + 255) / 256, 256, 0, stream>>>(vol, mask);

  const int vtiles = RPAD / 16;                          // 1189 m-tiles
  k_value<<<(vtiles + 3) / 4, 128, 0, stream>>>(f0, f1, f2, WvS, bv, value);

  const int qtiles = (BATCH * NTOT) / 16;                // 16384, exact
  k_fused<<<qtiles / 2, 64, 0, stream>>>(se, sp, refp, boff, bwgt, bo, lng, lnb,
                                         value, WcatS, WoS, mask, out);
}
